// SelfAttention_36687610643151
// MI455X (gfx1250) — matrix operand order repaired and checked by reading, not compiled, not run on hardware
//
#include <hip/hip_runtime.h>
#include <cstdint>
#include <cstddef>

#define DIMN 1536
#define NH   12
#define HD   128
#define SEQ  3072
#define BLKT 128
#define NB   24
#define TOPK 128
#define QKSCALE 0.08838834764831845f  /* 1/sqrt(128) */

typedef __attribute__((ext_vector_type(16))) __bf16 v16bf;
typedef __attribute__((ext_vector_type(8)))  __bf16 v8bf;
typedef __attribute__((ext_vector_type(8)))  float  v8f;

union ABFrag { v16bf v; v8bf h[2]; };

__device__ __forceinline__ v8f wmma_bf16(v16bf a, v16bf b, v8f c) {
  // D = A(16x32 bf16) x B(32x16 bf16) + C(16x16 f32)
  return __builtin_amdgcn_wmma_f32_16x16x32_bf16(false, a, false, b, (short)0, c,
                                                 false, false);
}

__device__ __forceinline__ float hmax16(float v) {
#pragma unroll
  for (int m = 1; m <= 8; m <<= 1) v = fmaxf(v, __shfl_xor(v, m, 32));
  return v;
}
__device__ __forceinline__ float hsum16(float v) {
#pragma unroll
  for (int m = 1; m <= 8; m <<= 1) v += __shfl_xor(v, m, 32);
  return v;
}

// ---------------------------------------------------------------- converts
__global__ void k_f32_to_bf16(const float* __restrict__ in,
                              __bf16* __restrict__ out, int n) {
  int i = blockIdx.x * 256 + threadIdx.x;
  if (i < n) out[i] = (__bf16)in[i];
}

// W [R,C] fp32 -> WT [C,R] bf16
__global__ void k_transpose_bf16(const float* __restrict__ W,
                                 __bf16* __restrict__ WT, int R, int C) {
  __shared__ float tile[32][33];
  int c0 = blockIdx.x * 32, r0 = blockIdx.y * 32;
  int x = threadIdx.x, y = threadIdx.y;  // (32,8)
#pragma unroll
  for (int i = 0; i < 4; ++i)
    tile[y + 8 * i][x] = W[(size_t)(r0 + y + 8 * i) * C + c0 + x];
  __syncthreads();
#pragma unroll
  for (int i = 0; i < 4; ++i)
    WT[(size_t)(c0 + y + 8 * i) * R + r0 + x] = (__bf16)tile[x][y + 8 * i];
}

// ---------------------------------------------------------------- GEMM
// C[M,N] = A[M,K](bf16) * BT[K,N](bf16, pre-transposed weights) + bias
// block tile 128x64, 8 waves of 32x32, K-steps of 32.
__global__ __launch_bounds__(256) void k_gemm_bf16(
    const __bf16* __restrict__ A, const __bf16* __restrict__ BT,
    const float* __restrict__ bias, float* __restrict__ Cf,
    __bf16* __restrict__ Cb, int M, int N, int K) {
  const int lane = threadIdx.x & 31, wave = threadIdx.x >> 5;
  const int l15 = lane & 15, hs = (lane >> 4) & 1, kofs = hs ? 8 : 0;
  const int m0 = blockIdx.x * 128 + (wave >> 1) * 32;
  const int n0 = blockIdx.y * 64 + (wave & 1) * 32;
  const v8f VZ = {0.f, 0.f, 0.f, 0.f, 0.f, 0.f, 0.f, 0.f};
  v8f acc[2][2];
#pragma unroll
  for (int i = 0; i < 2; ++i)
#pragma unroll
    for (int j = 0; j < 2; ++j) acc[i][j] = VZ;

  for (int k0 = 0; k0 < K; k0 += 32) {
    ABFrag a[2];
    v16bf b[2];
#pragma unroll
    for (int ti = 0; ti < 2; ++ti) {
      const __bf16* pa = A + (size_t)(m0 + ti * 16 + l15) * K + k0 + kofs;
      a[ti].h[0] = *(const v8bf*)pa;
      a[ti].h[1] = *(const v8bf*)(pa + 16);
    }
#pragma unroll
    for (int tj = 0; tj < 2; ++tj)
#pragma unroll
      for (int s = 0; s < 16; ++s)
        b[tj][s] = BT[(size_t)(k0 + kofs + (s & 7) + ((s >> 3) << 4)) * N + n0 + tj * 16 + l15];
    if (k0 + 32 < K) {
      __builtin_prefetch(BT + (size_t)(k0 + 32 + lane) * N + n0, 0, 3);
      __builtin_prefetch(A + (size_t)(m0 + l15) * K + k0 + 32, 0, 3);
    }
#pragma unroll
    for (int ti = 0; ti < 2; ++ti)
#pragma unroll
      for (int tj = 0; tj < 2; ++tj)
        acc[ti][tj] = wmma_bf16(a[ti].v, b[tj], acc[ti][tj]);
  }

#pragma unroll
  for (int ti = 0; ti < 2; ++ti)
#pragma unroll
    for (int tj = 0; tj < 2; ++tj)
#pragma unroll
      for (int r = 0; r < 8; ++r) {
        int row = m0 + ti * 16 + r + hs * 8;
        int col = n0 + tj * 16 + l15;
        float v = acc[ti][tj][r] + bias[col];
        if (Cf) Cf[(size_t)row * N + col] = v;
        if (Cb) Cb[(size_t)row * N + col] = (__bf16)v;
      }
}

// ---------------------------------------------------------------- norm+rope
// In-place RMSNorm over DIMN then RoPE; emits bf16 row-major (Q) or
// head-transposed [H*HD, S] (K) copies for the WMMA attention fragments.
__global__ __launch_bounds__(256) void k_rmsnorm_rope(
    float* __restrict__ Y, const float* __restrict__ g,
    const float* __restrict__ cs, const float* __restrict__ sn,
    __bf16* __restrict__ out_row, __bf16* __restrict__ out_tr) {
  const int t = blockIdx.x, tid = threadIdx.x;
  float xr[3], xi[3];
  float ss = 0.f;
#pragma unroll
  for (int kk = 0; kk < 3; ++kk) {
    int p = tid + 256 * kk;
    xr[kk] = Y[(size_t)t * DIMN + 2 * p];
    xi[kk] = Y[(size_t)t * DIMN + 2 * p + 1];
    ss += xr[kk] * xr[kk] + xi[kk] * xi[kk];
  }
  __shared__ float red[256];
  red[tid] = ss;
  __syncthreads();
  for (int s = 128; s > 0; s >>= 1) {
    if (tid < s) red[tid] += red[tid + s];
    __syncthreads();
  }
  const float inv = rsqrtf(red[0] * (1.f / DIMN) + 1e-5f);
#pragma unroll
  for (int kk = 0; kk < 3; ++kk) {
    int p = tid + 256 * kk, d = 2 * p;
    float yr = xr[kk] * inv * g[d];
    float yi = xi[kk] * inv * g[d + 1];
    float c = cs[t * (HD / 2) + (p & 63)];
    float s_ = sn[t * (HD / 2) + (p & 63)];
    float zr = yr * c - yi * s_;
    float zi = yr * s_ + yi * c;
    Y[(size_t)t * DIMN + d] = zr;
    Y[(size_t)t * DIMN + d + 1] = zi;
    if (out_row) {
      out_row[(size_t)t * DIMN + d] = (__bf16)zr;
      out_row[(size_t)t * DIMN + d + 1] = (__bf16)zi;
    }
    if (out_tr) {  // (h*HD + d_in_head) == d, so KT index is d*SEQ + t
      out_tr[(size_t)d * SEQ + t] = (__bf16)zr;
      out_tr[(size_t)(d + 1) * SEQ + t] = (__bf16)zi;
    }
  }
}

// ---------------------------------------------------------------- pooling
__global__ void k_pool(const float* __restrict__ Q, float* __restrict__ QW) {
  int nb = blockIdx.x;
  for (int d = threadIdx.x; d < DIMN; d += 256) {
    float s = 0.f;
    for (int tt = 0; tt < BLKT; ++tt) s += Q[(size_t)(nb * BLKT + tt) * DIMN + d];
    QW[nb * DIMN + d] = s * (1.f / BLKT);
  }
}

// ---------------------------------------------------------------- draft mask
__device__ __forceinline__ bool local_ok(int i, int j) {
  // 6x4 block grid, window 6 (half = 3): rows [r-3, r+2], cols [c-3, c+2]
  int ry = i >> 2, cx = i & 3, qy = j >> 2, qx = j & 3;
  return (qy >= ry - 3) && (qy <= ry + 2) && (qx >= cx - 3) && (qx <= cx + 2);
}

__global__ __launch_bounds__(256) void k_draft_mask(const float* __restrict__ QW,
                                                    const float* __restrict__ KW,
                                                    int* __restrict__ BM) {
  const int h = blockIdx.x, tid = threadIdx.x;
  __shared__ float attn[NB * NB];
  __shared__ float work[NB * NB];
  __shared__ float mval[256];
  __shared__ int marg[256];
  __shared__ float s_thr;
  for (int idx = tid; idx < NB * NB; idx += 256) {
    int i = idx / NB, j = idx % NB;
    float sv = -3.0e38f;
    if (local_ok(i, j)) {
      float acc = 0.f;
      const float* qp = QW + (size_t)i * DIMN + h * HD;
      const float* kp = KW + (size_t)j * DIMN + h * HD;
      for (int d = 0; d < HD; ++d) acc += qp[d] * kp[d];
      sv = acc * QKSCALE;
    }
    attn[idx] = sv;
  }
  __syncthreads();
  if (tid < NB) {  // row softmax (self block always unmasked -> finite max)
    float mx = -3.0e38f;
    for (int j = 0; j < NB; ++j) mx = fmaxf(mx, attn[tid * NB + j]);
    float s = 0.f;
    for (int j = 0; j < NB; ++j) {
      float v = attn[tid * NB + j];
      float e = (v > -1.0e37f) ? __expf(v - mx) : 0.f;
      attn[tid * NB + j] = e;
      s += e;
    }
    float is = 1.f / s;
    for (int j = 0; j < NB; ++j) attn[tid * NB + j] *= is;
  }
  __syncthreads();
  for (int idx = tid; idx < NB * NB; idx += 256) work[idx] = attn[idx];
  __syncthreads();
  // extract the (TOPK+1)-th largest value as threshold
  for (int it = 0; it <= TOPK; ++it) {
    float bv = -3.0e38f;
    int ba = -1;
    for (int idx = tid; idx < NB * NB; idx += 256)
      if (work[idx] > bv) { bv = work[idx]; ba = idx; }
    mval[tid] = bv;
    marg[tid] = ba;
    __syncthreads();
    if (tid == 0) {
      float b = -3.0e38f;
      int a = -1;
      for (int t2 = 0; t2 < 256; ++t2)
        if (mval[t2] > b) { b = mval[t2]; a = marg[t2]; }
      if (a >= 0) work[a] = -3.0e38f;
      if (it == TOPK) s_thr = b;
    }
    __syncthreads();
  }
  const float thr = s_thr;
  for (int idx = tid; idx < NB * NB; idx += 256)
    BM[(size_t)h * NB * NB + idx] = (attn[idx] > thr) ? 1 : 0;
}

// ---------------------------------------------------------------- attention
// One block per (query-block, head). 8 waves x 16 query rows.
// Streams only key blocks selected by the draft mask; online softmax.
__global__ __launch_bounds__(256) void k_attention(
    const __bf16* __restrict__ Qb, const __bf16* __restrict__ Kt,
    const __bf16* __restrict__ Vb, const int* __restrict__ BM,
    __bf16* __restrict__ Ob) {
  const int qb = blockIdx.x, h = blockIdx.y;
  const int lane = threadIdx.x & 31, wave = threadIdx.x >> 5;
  const int l15 = lane & 15, hs = (lane >> 4) & 1, kofs = hs ? 8 : 0;
  __shared__ __bf16 P[128][136];  // P tile staging (D-layout -> A-layout)
  const v8f VZ = {0.f, 0.f, 0.f, 0.f, 0.f, 0.f, 0.f, 0.f};

  // Q A-fragments for this wave's 16 rows, all of HD (4 K-steps of 32)
  ABFrag qf[4];
  const int rowA = qb * BLKT + wave * 16 + l15;
#pragma unroll
  for (int ks = 0; ks < 4; ++ks) {
    const __bf16* pq = Qb + (size_t)rowA * DIMN + h * HD + ks * 32 + kofs;
    qf[ks].h[0] = *(const v8bf*)pq;
    qf[ks].h[1] = *(const v8bf*)(pq + 16);
  }

  v8f oacc[8];
#pragma unroll
  for (int j = 0; j < 8; ++j) oacc[j] = VZ;
  float mrow[8], lrow[8];
#pragma unroll
  for (int r = 0; r < 8; ++r) { mrow[r] = -3.0e38f; lrow[r] = 0.f; }

  for (int kb = 0; kb < NB; ++kb) {
    if (!BM[(h * NB + qb) * NB + kb]) continue;  // uniform across block

    // S = Q * K^T  (Kt pre-transposed: contiguous B-fragment loads)
    v8f sacc[8];
#pragma unroll
    for (int j = 0; j < 8; ++j) sacc[j] = VZ;
    for (int ks = 0; ks < 4; ++ks) {
      const __bf16* pk = Kt + (size_t)(h * HD + ks * 32) * SEQ + kb * BLKT + l15;
#pragma unroll
      for (int j = 0; j < 8; ++j) {
        v16bf b;
#pragma unroll
        for (int s = 0; s < 16; ++s)
          b[s] = pk[(size_t)(kofs + (s & 7) + ((s >> 3) << 4)) * SEQ + j * 16];
        sacc[j] = wmma_bf16(qf[ks].v, b, sacc[j]);
      }
    }

    // online softmax update (row stats across 16 lanes of each half-wave)
    float alpha[8];
#pragma unroll
    for (int r = 0; r < 8; ++r) {
      float mx = -3.0e38f;
#pragma unroll
      for (int j = 0; j < 8; ++j) mx = fmaxf(mx, sacc[j][r]);
      mx = hmax16(mx * QKSCALE);
      float mn = fmaxf(mrow[r], mx);
      alpha[r] = __expf(mrow[r] - mn);
      mrow[r] = mn;
    }

    __syncthreads();  // protect P against previous iteration's readers
#pragma unroll
    for (int r = 0; r < 8; ++r) {
      int prow = wave * 16 + r + hs * 8;
      float rs = 0.f;
#pragma unroll
      for (int j = 0; j < 8; ++j) {
        float pv = __expf(sacc[j][r] * QKSCALE - mrow[r]);
        rs += pv;
        P[prow][j * 16 + l15] = (__bf16)pv;
      }
      rs = hsum16(rs);
      lrow[r] = lrow[r] * alpha[r] + rs;
#pragma unroll
      for (int j2 = 0; j2 < 8; ++j2) oacc[j2][r] *= alpha[r];
    }
    __syncthreads();

    // O += P * V   (V row-major is already the K x N B-layout)
    for (int ks = 0; ks < 4; ++ks) {
      ABFrag pa;
      const __bf16* pl = &P[wave * 16 + l15][ks * 32 + kofs];
      pa.h[0] = *(const v8bf*)pl;
      pa.h[1] = *(const v8bf*)(pl + 16);
      const __bf16* pv = Vb + (size_t)(kb * BLKT + ks * 32) * DIMN + h * HD + l15;
#pragma unroll
      for (int j2 = 0; j2 < 8; ++j2) {
        v16bf b;
#pragma unroll
        for (int s = 0; s < 16; ++s)
          b[s] = pv[(size_t)(kofs + (s & 7) + ((s >> 3) << 4)) * DIMN + j2 * 16];
        oacc[j2] = wmma_bf16(pa.v, b, oacc[j2]);
      }
    }
  }

#pragma unroll
  for (int j2 = 0; j2 < 8; ++j2)
#pragma unroll
    for (int r = 0; r < 8; ++r) {
      int row = qb * BLKT + wave * 16 + r + hs * 8;
      float den = lrow[r] > 0.f ? lrow[r] : 1.f;
      Ob[(size_t)row * DIMN + h * HD + j2 * 16 + l15] =
          (__bf16)(oacc[j2][r] / den);
    }
}

// ---------------------------------------------------------------- launch
extern "C" void kernel_launch(void* const* d_in, const int* in_sizes, int n_in,
                              void* d_out, int out_size, void* d_ws,
                              size_t ws_size, hipStream_t stream) {
  const float* x = (const float*)d_in[0];
  const float* wq = (const float*)d_in[1];
  const float* bq = (const float*)d_in[2];
  const float* wk = (const float*)d_in[3];
  const float* bk = (const float*)d_in[4];
  const float* wv = (const float*)d_in[5];
  const float* bv = (const float*)d_in[6];
  const float* wo = (const float*)d_in[7];
  const float* bo = (const float*)d_in[8];
  const float* gq = (const float*)d_in[9];
  const float* gk = (const float*)d_in[10];
  const float* fcs = (const float*)d_in[11];
  const float* fsn = (const float*)d_in[12];
  float* out = (float*)d_out;
  (void)in_sizes; (void)n_in; (void)out_size; (void)ws_size;

  char* ws = (char*)d_ws;
  size_t off = 0;
  auto take = [&](size_t bytes) {
    char* p = ws + off;
    off = (off + bytes + 255) & ~(size_t)255;
    return p;
  };
  __bf16* xbf = (__bf16*)take((size_t)SEQ * DIMN * 2);
  __bf16* wqT = (__bf16*)take((size_t)DIMN * DIMN * 2);
  __bf16* wkT = (__bf16*)take((size_t)DIMN * DIMN * 2);
  __bf16* wvT = (__bf16*)take((size_t)DIMN * DIMN * 2);
  __bf16* woT = (__bf16*)take((size_t)DIMN * DIMN * 2);
  float* qf = (float*)take((size_t)SEQ * DIMN * 4);
  float* kf = (float*)take((size_t)SEQ * DIMN * 4);
  __bf16* qbf = (__bf16*)take((size_t)SEQ * DIMN * 2);
  __bf16* ktbf = (__bf16*)take((size_t)SEQ * DIMN * 2);
  __bf16* vbf = (__bf16*)take((size_t)SEQ * DIMN * 2);
  __bf16* obf = (__bf16*)take((size_t)SEQ * DIMN * 2);
  float* qw = (float*)take((size_t)NB * DIMN * 4);
  float* kw = (float*)take((size_t)NB * DIMN * 4);
  int* bm = (int*)take((size_t)NH * NB * NB * 4);

  k_f32_to_bf16<<<(SEQ * DIMN + 255) / 256, 256, 0, stream>>>(x, xbf, SEQ * DIMN);

  dim3 tg(DIMN / 32, DIMN / 32), tb(32, 8);
  k_transpose_bf16<<<tg, tb, 0, stream>>>(wq, wqT, DIMN, DIMN);
  k_transpose_bf16<<<tg, tb, 0, stream>>>(wk, wkT, DIMN, DIMN);
  k_transpose_bf16<<<tg, tb, 0, stream>>>(wv, wvT, DIMN, DIMN);
  k_transpose_bf16<<<tg, tb, 0, stream>>>(wo, woT, DIMN, DIMN);

  dim3 gg(SEQ / 128, DIMN / 64);
  k_gemm_bf16<<<gg, 256, 0, stream>>>(xbf, wqT, bq, qf, (__bf16*)nullptr, SEQ,
                                      DIMN, DIMN);
  k_gemm_bf16<<<gg, 256, 0, stream>>>(xbf, wkT, bk, kf, (__bf16*)nullptr, SEQ,
                                      DIMN, DIMN);
  k_gemm_bf16<<<gg, 256, 0, stream>>>(xbf, wvT, bv, (float*)nullptr, vbf, SEQ,
                                      DIMN, DIMN);

  k_rmsnorm_rope<<<SEQ, 256, 0, stream>>>(qf, gq, fcs, fsn, qbf,
                                          (__bf16*)nullptr);
  k_rmsnorm_rope<<<SEQ, 256, 0, stream>>>(kf, gk, fcs, fsn, (__bf16*)nullptr,
                                          ktbf);

  k_pool<<<NB, 256, 0, stream>>>(qf, qw);
  k_pool<<<NB, 256, 0, stream>>>(kf, kw);
  k_draft_mask<<<NH, 256, 0, stream>>>(qw, kw, bm);

  k_attention<<<dim3(NB, NH), 256, 0, stream>>>(qbf, ktbf, vbf, bm, obf);

  k_gemm_bf16<<<gg, 256, 0, stream>>>(obf, woT, bo, out, (__bf16*)nullptr, SEQ,
                                      DIMN, DIMN);
}
